// TokenGTEncoder_47072841564361
// MI455X (gfx1250) — matrix operand order repaired and checked by reading, not compiled, not run on hardware
//
#include <hip/hip_runtime.h>
#include <math.h>

// ---------------------------------------------------------------------------
// Problem constants (fixed by the reference harness)
// ---------------------------------------------------------------------------
static constexpr int Bsz  = 32;           // batch
static constexpr int T    = 256;          // tokens
static constexpr int D    = 1024;         // hidden
static constexpr int NPER = 64;           // atoms per sample
static constexpr int Nnod = Bsz * NPER;   // 2048 nodes
static constexpr int E    = 65536;        // edges
static constexpr int K2   = 2 * D;        // 2048 (concat dim)

// Force-head tiling
static constexpr int SLICES  = 8;         // column slices (128 cols each)
static constexpr int NSLICE  = 128;       // columns per workgroup
static constexpr int MWG     = 256;       // edges per workgroup (8 waves x 32)
static constexpr int KC      = 64;        // K elements per LDS chunk
static constexpr int CHUNKS  = K2 / KC;   // 32
static constexpr int BST     = KC + 8;    // padded LDS row stride (bank-safe)

typedef __attribute__((ext_vector_type(16))) __bf16 v16bf;
typedef __attribute__((ext_vector_type(8)))  __bf16 v8bf;
typedef __attribute__((ext_vector_type(8)))  float  v8f;

union BF16x16 { v16bf v; v8bf h[2]; };

__device__ __forceinline__ float gelu_exact(float x) {
    return 0.5f * x * (1.0f + erff(x * 0.70710678118654752440f));
}

// ---------------------------------------------------------------------------
// gfx1250 async global->LDS copy (ASYNCcnt path)
// ---------------------------------------------------------------------------
__device__ __forceinline__ void async_copy_b128(void* lds_dst, const void* gsrc) {
    // Low 32 bits of a generic pointer into LDS hold the LDS byte address.
    unsigned lds_off = (unsigned)(unsigned long long)lds_dst;
    asm volatile("global_load_async_to_lds_b128 %0, %1, off"
                 :: "v"(lds_off), "v"(gsrc) : "memory");
}

__device__ __forceinline__ void wait_async_le4() {
#if __has_builtin(__builtin_amdgcn_s_wait_asynccnt)
    __builtin_amdgcn_s_wait_asynccnt(4);
#else
    asm volatile("s_wait_asynccnt 0x4" ::: "memory");
#endif
}
__device__ __forceinline__ void wait_async_le0() {
#if __has_builtin(__builtin_amdgcn_s_wait_asynccnt)
    __builtin_amdgcn_s_wait_asynccnt(0);
#else
    asm volatile("s_wait_asynccnt 0x0" ::: "memory");
#endif
}

// ---------------------------------------------------------------------------
// Zero the output buffer (forces are accumulated with atomics)
// ---------------------------------------------------------------------------
__global__ void zero_out_kernel(float* __restrict__ out, int n) {
    int i = blockIdx.x * blockDim.x + threadIdx.x;
    if (i < n) out[i] = 0.0f;
}

// ---------------------------------------------------------------------------
// Convert node features x[:,1:1+NPER,:] -> bf16, row-major [Nnod][D]
// ---------------------------------------------------------------------------
__global__ void cvt_nodes_kernel(const float* __restrict__ x,
                                 __bf16* __restrict__ nodesB) {
    int idx = blockIdx.x * blockDim.x + threadIdx.x;   // Nnod*D elements
    int n = idx >> 10;
    int k = idx & (D - 1);
    int b = n >> 6;
    int i = n & (NPER - 1);
    size_t src = ((size_t)(b * T + 1 + i) << 10) + k;
    nodesB[idx] = (__bf16)x[src];
}

// ---------------------------------------------------------------------------
// Convert Wf1 (K2 x D, row-major) -> transposed bf16 WfT [D][K2]
// ---------------------------------------------------------------------------
__global__ void cvt_wf1_kernel(const float* __restrict__ Wf1,
                               __bf16* __restrict__ WfT) {
    int idx = blockIdx.x * blockDim.x + threadIdx.x;   // D*K2 elements
    int j = idx >> 11;
    int k = idx & (K2 - 1);
    WfT[idx] = (__bf16)Wf1[(size_t)k * D + j];
}

// ---------------------------------------------------------------------------
// Energy head: LN -> GEMM(32x1024x1024) + GELU -> dot(We2) + be2
// ---------------------------------------------------------------------------
__global__ __launch_bounds__(256)
void energy_kernel(const float* __restrict__ x,
                   const float* __restrict__ ln_g, const float* __restrict__ ln_b,
                   const float* __restrict__ We1,  const float* __restrict__ be1,
                   const float* __restrict__ We2,  const float* __restrict__ be2,
                   float* __restrict__ energy) {
    __shared__ float sred[256];
    __shared__ float hn[D];
    const int b   = blockIdx.x;
    const int tid = threadIdx.x;
    const float* xr = x + (size_t)b * T * D;

    float s = 0.f;
    for (int k = tid; k < D; k += 256) s += xr[k];
    sred[tid] = s; __syncthreads();
    for (int off = 128; off > 0; off >>= 1) {
        if (tid < off) sred[tid] += sred[tid + off];
        __syncthreads();
    }
    float mu = sred[0] * (1.0f / D);
    __syncthreads();

    float v = 0.f;
    for (int k = tid; k < D; k += 256) { float d = xr[k] - mu; v += d * d; }
    sred[tid] = v; __syncthreads();
    for (int off = 128; off > 0; off >>= 1) {
        if (tid < off) sred[tid] += sred[tid + off];
        __syncthreads();
    }
    float rstd = rsqrtf(sred[0] * (1.0f / D) + 1e-5f);
    __syncthreads();

    for (int k = tid; k < D; k += 256)
        hn[k] = (xr[k] - mu) * rstd * ln_g[k] + ln_b[k];
    __syncthreads();

    float esum = 0.f;
    for (int j = tid; j < D; j += 256) {
        float acc = be1[j];
        for (int k = 0; k < D; ++k)
            acc = fmaf(hn[k], We1[(size_t)k * D + j], acc);
        esum += gelu_exact(acc) * We2[j];
    }
    sred[tid] = esum; __syncthreads();
    for (int off = 128; off > 0; off >>= 1) {
        if (tid < off) sred[tid] += sred[tid + off];
        __syncthreads();
    }
    if (tid == 0) energy[b] = sred[0] + be2[0];
}

// ---------------------------------------------------------------------------
// Force head, sliced bf16 WMMA GEMM:
//   grid = (E/MWG edge blocks, SLICES column slices), 256 threads (8 waves).
//   WG computes partial fm over its 128-column slice for 256 edges:
//     fm_s[m] = sum_{j in slice} gelu(A[m]@Wf1[:,j] + bf1[j]) * Wf2[j]
//   and atomically adds fm_s * vec/dist into the forces (linear in slices).
//   B (WfT) is streamed through double-buffered LDS chunks with
//   global_load_async_to_lds_b128, shared by all 8 waves.
// ---------------------------------------------------------------------------
__global__ __launch_bounds__(256)
void edge_gemm_kernel(const __bf16* __restrict__ nodesB,
                      const __bf16* __restrict__ WfT,
                      const int*    __restrict__ ei0,
                      const int*    __restrict__ ei1,
                      const float*  __restrict__ bf1,
                      const float*  __restrict__ Wf2,
                      const float*  __restrict__ bf2,
                      const float*  __restrict__ evec,
                      const float*  __restrict__ edist,
                      float* __restrict__ forces) {
    __shared__ __bf16 Bs[2][NSLICE * BST];   // 2 x 18 KiB (padded)
    __shared__ float  fmbuf[8][32];

    const int eb    = blockIdx.x;            // edge block (256 edges)
    const int slice = blockIdx.y;            // column slice (128 cols)
    const int tid   = threadIdx.x;
    const int w     = tid >> 5;              // wave 0..7
    const int lane  = tid & 31;
    const int r     = lane & 15;
    const int hi    = lane >> 4;
    const int sel8  = hi * 8;

    // This wave's 32 edges
    const int m0 = eb * MWG + w * 32 + r;    // M-tile 0 row
    const int m1 = m0 + 16;                  // M-tile 1 row
    const size_t n00 = (size_t)ei0[m0], n01 = (size_t)ei1[m0];
    const size_t n10 = (size_t)ei0[m1], n11 = (size_t)ei1[m1];

    v8f acc[2][8];
#pragma unroll
    for (int t = 0; t < 2; ++t)
#pragma unroll
        for (int u = 0; u < 8; ++u)
#pragma unroll
            for (int vv = 0; vv < 8; ++vv) acc[t][u][vv] = 0.0f;

    const __bf16* wslice = WfT + (size_t)(slice * NSLICE) * K2;

    // Async-stage one K-chunk of B (NSLICE x KC bf16 = 16 KiB) into buffer `bb`.
    // 1024 x 16B transfers; 4 per thread -> 4 async instructions per wave.
    auto stage_chunk = [&](int ch, int bb) {
        const int kbase = ch * KC;
#pragma unroll
        for (int q = 0; q < 4; ++q) {
            int id = q * 256 + tid;          // 0..1023
            int c  = id >> 3;                // column in slice 0..127
            int ko = (id & 7) * 8;           // 0,8,...,56
            const __bf16* src = wslice + (size_t)c * K2 + kbase + ko;
            async_copy_b128(&Bs[bb][c * BST + ko], src);
        }
    };

    stage_chunk(0, 0);
    int buf = 0;
    for (int ch = 0; ch < CHUNKS; ++ch) {
        const bool more = (ch + 1) < CHUNKS;
        if (more) stage_chunk(ch + 1, buf ^ 1);
        if (more) wait_async_le4(); else wait_async_le0();
        __syncthreads();                      // chunk `ch` visible to all waves

        const int kbase = ch * KC;
        const int kloc  = kbase & (D - 1);    // offset inside node row
        const bool ph1  = kbase >= D;         // second half: ei1 nodes
        const __bf16* a0 = nodesB + (ph1 ? n01 : n00) * D + kloc;
        const __bf16* a1 = nodesB + (ph1 ? n11 : n10) * D + kloc;

#pragma unroll
        for (int ks = 0; ks < KC; ks += 32) {
            BF16x16 A0, A1;
            A0.h[0] = *(const v8bf*)(a0 + ks + sel8);
            A0.h[1] = *(const v8bf*)(a0 + ks + 16 + sel8);
            A1.h[0] = *(const v8bf*)(a1 + ks + sel8);
            A1.h[1] = *(const v8bf*)(a1 + ks + 16 + sel8);
#pragma unroll
            for (int u = 0; u < 8; ++u) {
                const __bf16* bp = &Bs[buf][(u * 16 + r) * BST + ks + sel8];
                BF16x16 Bf;
                Bf.h[0] = *(const v8bf*)(bp);
                Bf.h[1] = *(const v8bf*)(bp + 16);
                acc[0][u] = __builtin_amdgcn_wmma_f32_16x16x32_bf16(
                    false, A0.v, false, Bf.v, (short)0, acc[0][u], false, false);
                acc[1][u] = __builtin_amdgcn_wmma_f32_16x16x32_bf16(
                    false, A1.v, false, Bf.v, (short)0, acc[1][u], false, false);
            }
        }
        __syncthreads();                      // done reading before overwrite
        buf ^= 1;
    }

    // Epilogue: partial fm over this slice's columns
    float srow[2][8];
#pragma unroll
    for (int t = 0; t < 2; ++t)
#pragma unroll
        for (int vv = 0; vv < 8; ++vv) srow[t][vv] = 0.0f;

#pragma unroll
    for (int u = 0; u < 8; ++u) {
        const int c    = slice * NSLICE + u * 16 + r;
        const float b1 = bf1[c];
        const float w2 = Wf2[c];
#pragma unroll
        for (int vv = 0; vv < 8; ++vv) {
            srow[0][vv] += gelu_exact(acc[0][u][vv] + b1) * w2;
            srow[1][vv] += gelu_exact(acc[1][u][vv] + b1) * w2;
        }
    }

    // Reduce across the 16 lanes of each half (the wave's 128 columns)
#pragma unroll
    for (int mask = 1; mask < 16; mask <<= 1) {
#pragma unroll
        for (int t = 0; t < 2; ++t)
#pragma unroll
            for (int vv = 0; vv < 8; ++vv)
                srow[t][vv] += __shfl_xor(srow[t][vv], mask, 32);
    }

    if (r == 0) {                             // lanes 0 and 16 hold the sums
#pragma unroll
        for (int t = 0; t < 2; ++t)
#pragma unroll
            for (int vv = 0; vv < 8; ++vv)
                fmbuf[w][t * 16 + hi * 8 + vv] = srow[t][vv];
    }
    __syncthreads();

    // One thread per (wave, row): scatter partial forces
    {
        const int wv  = tid >> 5;
        const int row = tid & 31;
        float fm = fmbuf[wv][row] + bf2[0] * (1.0f / SLICES);
        const int m    = eb * MWG + wv * 32 + row;
        const float cf = fm / edist[m];
        const int dst  = ei0[m] * 3;
        atomicAdd(&forces[dst + 0], cf * evec[m * 3 + 0]);
        atomicAdd(&forces[dst + 1], cf * evec[m * 3 + 1]);
        atomicAdd(&forces[dst + 2], cf * evec[m * 3 + 2]);
    }
}

// ---------------------------------------------------------------------------
// Host-side launcher
// ---------------------------------------------------------------------------
extern "C" void kernel_launch(void* const* d_in, const int* in_sizes, int n_in,
                              void* d_out, int out_size, void* d_ws, size_t ws_size,
                              hipStream_t stream) {
    const float* x     = (const float*)d_in[0];
    const int*   eidx  = (const int*)  d_in[2];
    const float* evec  = (const float*)d_in[3];
    const float* edist = (const float*)d_in[4];
    const float* ln_g  = (const float*)d_in[6];
    const float* ln_b  = (const float*)d_in[7];
    const float* We1   = (const float*)d_in[8];
    const float* be1   = (const float*)d_in[9];
    const float* We2   = (const float*)d_in[10];
    const float* be2   = (const float*)d_in[11];
    const float* Wf1   = (const float*)d_in[12];
    const float* bf1   = (const float*)d_in[13];
    const float* Wf2   = (const float*)d_in[14];
    const float* bf2   = (const float*)d_in[15];

    float* out    = (float*)d_out;
    float* energy = out;          // 32
    float* forces = out + Bsz;    // 2048*3

    __bf16* nodesB = (__bf16*)d_ws;                                 // 4 MB
    __bf16* WfT    = (__bf16*)((char*)d_ws + (size_t)Nnod * D * 2); // 4 MB

    const int* ei0 = eidx;
    const int* ei1 = eidx + E;

    {
        const int n = Bsz + Nnod * 3;
        zero_out_kernel<<<(n + 255) / 256, 256, 0, stream>>>(out, n);
    }
    cvt_nodes_kernel<<<(Nnod * D) / 256, 256, 0, stream>>>(x, nodesB);
    cvt_wf1_kernel<<<(D * K2) / 256, 256, 0, stream>>>(Wf1, WfT);
    energy_kernel<<<Bsz, 256, 0, stream>>>(x, ln_g, ln_b, We1, be1, We2, be2, energy);

    dim3 grid(E / MWG, SLICES);
    edge_gemm_kernel<<<grid, 256, 0, stream>>>(nodesB, WfT, ei0, ei1,
                                               bf1, Wf2, bf2, evec, edist, forces);
}
